// Decoder_33294586479282
// MI455X (gfx1250) — matrix operand order repaired and checked by reading, not compiled, not run on hardware
//
#include <hip/hip_runtime.h>

// DA-RNN decoder for MI455X (gfx1250, wave32, WMMA).
// - enc @ W_e.T is time-invariant -> precomputed once with f16 WMMA.
// - enc_proj + enc kept as f16 (67MB each) so the 128-step scan streams from
//   the 192MB L2 instead of HBM (f32 would be 268MB and spill to HBM).
// - attn_b2 is a uniform shift over t -> cancels in softmax; dropped.
// - Recurrent kernel: 128 blocks x 256 threads (8 wave32 waves); each block
//   owns 8 independent batch rows for all 128 steps (no grid sync needed).
//   WMMA tiles are M=16 with 8 zero phantom rows (WMMA is ~2% of the work;
//   doubling block count doubles WGP coverage and the ~53KB LDS footprint
//   lets ~4 blocks co-reside per WGP to hide L2 latency in the tanh phase).

typedef __attribute__((ext_vector_type(16))) _Float16 v16h;
typedef __attribute__((ext_vector_type(8)))  _Float16 v8h;
typedef __attribute__((ext_vector_type(8)))  float    v8f;
typedef __attribute__((ext_vector_type(4)))  float    v4f;

#define DEV static __device__ __forceinline__

DEV float fast_tanh(float x){ float e = __expf(2.f*x); return 1.f - 2.f/(e+1.f); }
DEV float fast_sig(float x){ return 1.f/(1.f+__expf(-x)); }
DEV float wave_red_sum(float v){
  #pragma unroll
  for (int m = 16; m >= 1; m >>= 1) v += __shfl_xor(v, m, 32);
  return v;
}
DEV float wave_red_max(float v){
  #pragma unroll
  for (int m = 16; m >= 1; m >>= 1) v = fmaxf(v, __shfl_xor(v, m, 32));
  return v;
}

DEV v16h wmma_b_frag(const _Float16* brow, int kbb){
  union { v16h v; v8h h2[2]; } ub;
  ub.h2[0] = *(const v8h*)(brow + kbb - ((kbb & 16) >> 1));
  ub.h2[1] = *(const v8h*)(brow + kbb - ((kbb & 16) >> 1) + 16);
  return ub.v;
}

// ---------------------------------------------------------------------------
// Kernel 1: convert weights to f16 in WMMA-B-friendly layout (+ combined bias)
// ---------------------------------------------------------------------------
__global__ void prep_weights(const float* __restrict__ attn_w1,
                             const float* __restrict__ w_hh_f,
                             const float* __restrict__ b_ih,
                             const float* __restrict__ b_hh,
                             _Float16* __restrict__ w1hc,   // [256][512]
                             _Float16* __restrict__ w1e,    // [256][256]
                             _Float16* __restrict__ whh,    // [1024][256]
                             float*    __restrict__ biasc)  // [1024]
{
  int i = blockIdx.x * blockDim.x + threadIdx.x;
  const int N1 = 256*512, N2 = N1 + 256*256, N3 = N2 + 1024*256, N4 = N3 + 1024;
  if (i < N1) {
    int n = i >> 9, k = i & 511;
    w1hc[i] = (_Float16)attn_w1[n*768 + k];              // columns 0..511
  } else if (i < N2) {
    int j = i - N1; int n = j >> 8, k = j & 255;
    w1e[j] = (_Float16)attn_w1[n*768 + 512 + k];         // columns 512..767
  } else if (i < N3) {
    int j = i - N2;
    whh[j] = (_Float16)w_hh_f[j];
  } else if (i < N4) {
    int j = i - N3;
    biasc[j] = b_ih[j] + b_hh[j];
  }
}

// ---------------------------------------------------------------------------
// Kernel 2: enc_proj = enc @ W_e.T + b1  (M=131072,N=256,K=256) via WMMA f16,
// also emits enc_f16 copy. One 16x16 output tile per wave.
// ---------------------------------------------------------------------------
__global__ __launch_bounds__(256) void encp_kernel(
    const float*    __restrict__ enc,    // [131072][256] f32
    const float*    __restrict__ b1,     // [256]
    const _Float16* __restrict__ w1e,    // [256][256]
    _Float16*       __restrict__ encp,   // [131072][256] f16 out
    _Float16*       __restrict__ encf)   // [131072][256] f16 out
{
  __shared__ _Float16 stile[8][16][16];
  const int lane = threadIdx.x & 31;
  const int wave = threadIdx.x >> 5;
  const int tile = blockIdx.x * 8 + wave;
  const int mtile = tile >> 4;          // 0..8191
  const int ntile = tile & 15;          // 0..15
  const int hf = lane >> 4;
  const int ln = lane & 15;

  const long rowA = (long)mtile * 16 + ln;           // A-fragment row (m = ln)
  const float*    arow = enc  + rowA * 256;
  const _Float16* brow = w1e  + (ntile*16 + ln) * 256;
  _Float16*       frow = encf + rowA * 256;

  v8f acc;
  float bn = b1[ntile*16 + ln];
  #pragma unroll
  for (int r = 0; r < 8; ++r) acc[r] = bn;           // C column n = bias b1[n]

  #pragma unroll
  for (int kb = 0; kb < 8; ++kb) {
    // A: lane = row m, f16 elems 0..7 at k=kb*32+8*hf, 8..15 at k=kb*32+16+8*hf
    union { v16h v; v8h h2[2]; _Float16 e[16]; } ua;
    const int k1 = kb*32 + hf*8;
    const int k2 = kb*32 + 16 + hf*8;
    v4f f0 = *(const v4f*)(arow + k1), f1 = *(const v4f*)(arow + k1 + 4);
    v4f f2 = *(const v4f*)(arow + k2), f3 = *(const v4f*)(arow + k2 + 4);
    #pragma unroll
    for (int j = 0; j < 4; ++j) {
      ua.e[j]    = (_Float16)f0[j];  ua.e[4+j]  = (_Float16)f1[j];
      ua.e[8+j]  = (_Float16)f2[j];  ua.e[12+j] = (_Float16)f3[j];
    }
    if (ntile == 0) {                                // each row converted once
      *(v8h*)(frow + k1) = ua.h2[0];
      *(v8h*)(frow + k2) = ua.h2[1];
    }
    v16h bv = wmma_b_frag(brow, kb*32 + hf*16);
    acc = __builtin_amdgcn_wmma_f32_16x16x32_f16(false, ua.v, false, bv,
                                                 (short)0, acc, false, false);
  }
  // stage D (col n, rows r+8*hf) through LDS, store coalesced as f16
  #pragma unroll
  for (int r = 0; r < 8; ++r) stile[wave][r + 8*hf][ln] = (_Float16)acc[r];
  __syncthreads();
  const int srow = lane >> 1, scol = (lane & 1) * 8;
  v8h outv = *(v8h*)&stile[wave][srow][scol];
  *(v8h*)(encp + ((long)mtile*16 + srow)*256 + ntile*16 + scol) = outv;
}

// ---------------------------------------------------------------------------
// Kernel 3: persistent recurrent decoder. 128 blocks x 256 threads (8 waves);
// each block owns 8 batch rows for all 128 steps. WMMA uses M=16 tiles with
// rows 8..15 as zero phantoms (hcat rows 8..15 stay zero forever).
// ---------------------------------------------------------------------------
__global__ __launch_bounds__(256) void decoder_kernel(
    const float*    __restrict__ y_hist,  // [1024][128]
    const float*    __restrict__ aw2,     // [256]
    const float*    __restrict__ w_ih,    // [1024]
    const float*    __restrict__ fc_w,    // [257]
    const float*    __restrict__ fc_b,    // [1]
    const float*    __restrict__ fcf_w,   // [512]
    const float*    __restrict__ fcf_b,   // [1]
    const _Float16* __restrict__ w1hc,    // [256][512]
    const _Float16* __restrict__ whh,     // [1024][256]
    const float*    __restrict__ biasc,   // [1024]
    const _Float16* __restrict__ encp,    // [1024*128][256]
    const _Float16* __restrict__ encf,    // [1024*128][256]
    float*          __restrict__ out)     // [1024]
{
  __shared__ float    sh_h[8][256];
  __shared__ float    sh_c[8][256];
  __shared__ _Float16 sh_hcat[16][520];   // [h|c] f16, padded vs 64-bank LDS
  __shared__ float    sh_hp[8][256];
  __shared__ float    sh_sc[8][128];
  __shared__ float    sh_ctx[8][256];
  __shared__ float    sh_yt[16];          // rows 8..15 = phantom zeros

  const int tid  = threadIdx.x;
  const int lane = tid & 31;
  const int w    = tid >> 5;             // wave 0..7 <-> batch row w
  const int hf   = lane >> 4;
  const int ln   = lane & 15;
  const long rowb = (long)(blockIdx.x * 8 + w);   // this wave's batch row

  for (int i = tid; i < 8*256; i += 256) { (&sh_h[0][0])[i] = 0.f; (&sh_c[0][0])[i] = 0.f; }
  for (int i = tid; i < 16*520; i += 256) (&sh_hcat[0][0])[i] = (_Float16)0.f;
  if (tid < 16) sh_yt[tid] = 0.f;
  __syncthreads();

  // per-lane constants for the row-local phases (lane owns e = lane*8..+7)
  float w2r[8], fcwr[8];
  #pragma unroll
  for (int j = 0; j < 8; ++j) { w2r[j] = aw2[lane*8 + j]; fcwr[j] = fc_w[lane*8 + j]; }
  const float fcw_y = fc_w[256], fcb = fc_b[0];
  const _Float16* encp_row = encp + rowb * 128 * 256;
  const _Float16* encf_row = encf + rowb * 128 * 256;

  for (int step = 0; step < 128; ++step) {
    // --- hp = [h|c]_f16 @ W_hc.T (M=16,K=512,N=256); wave w -> N-tiles w,w+8
    {
      v8f a0, a1;
      #pragma unroll
      for (int r = 0; r < 8; ++r) { a0[r] = 0.f; a1[r] = 0.f; }
      const _Float16* brow0 = w1hc + ( w      *16 + ln) * 512;
      const _Float16* brow1 = w1hc + ((w + 8) *16 + ln) * 512;
      #pragma unroll 4
      for (int kb = 0; kb < 16; ++kb) {
        union { v16h v; v8h h2[2]; } ua;
        const int k1 = kb*32 + hf*8;
        ua.h2[0] = *(const v8h*)&sh_hcat[ln][k1];
        ua.h2[1] = *(const v8h*)&sh_hcat[ln][k1 + 16];
        const int kbb = kb*32 + hf*16;
        a0 = __builtin_amdgcn_wmma_f32_16x16x32_f16(false, ua.v, false,
              wmma_b_frag(brow0, kbb), (short)0, a0, false, false);
        a1 = __builtin_amdgcn_wmma_f32_16x16x32_f16(false, ua.v, false,
              wmma_b_frag(brow1, kbb), (short)0, a1, false, false);
      }
      if (hf == 0) {                       // rows 0..7 valid, 8..15 phantom
        #pragma unroll
        for (int r = 0; r < 8; ++r) {
          sh_hp[r][ w     *16 + ln] = a0[r];
          sh_hp[r][(w + 8)*16 + ln] = a1[r];
        }
      }
    }
    __syncthreads();

    // --- scores / softmax / context / y_tild: wave-local to row w -----------
    {
      float hp8[8];
      #pragma unroll
      for (int j = 0; j < 8; ++j) hp8[j] = sh_hp[w][lane*8 + j];
      #pragma unroll 2
      for (int t = 0; t < 128; ++t) {                 // 512B/iter from L2
        v8h ev = *(const v8h*)(encp_row + t*256 + lane*8);
        float s = 0.f;
        #pragma unroll
        for (int j = 0; j < 8; ++j) s += w2r[j] * fast_tanh((float)ev[j] + hp8[j]);
        s = wave_red_sum(s);
        if (lane == 0) sh_sc[w][t] = s;
      }
      float sv[4];
      #pragma unroll
      for (int j = 0; j < 4; ++j) sv[j] = sh_sc[w][lane + 32*j];
      float mx = wave_red_max(fmaxf(fmaxf(sv[0], sv[1]), fmaxf(sv[2], sv[3])));
      float sum = 0.f;
      #pragma unroll
      for (int j = 0; j < 4; ++j) { sv[j] = __expf(sv[j] - mx); sum += sv[j]; }
      sum = wave_red_sum(sum);
      const float inv = 1.f / sum;
      #pragma unroll
      for (int j = 0; j < 4; ++j) sh_sc[w][lane + 32*j] = sv[j] * inv;

      float cacc[8];
      #pragma unroll
      for (int j = 0; j < 8; ++j) cacc[j] = 0.f;
      #pragma unroll 4
      for (int t = 0; t < 128; ++t) {
        const float a = sh_sc[w][t];                  // LDS broadcast
        v8h ev = *(const v8h*)(encf_row + t*256 + lane*8);
        #pragma unroll
        for (int j = 0; j < 8; ++j) cacc[j] += a * (float)ev[j];
      }
      #pragma unroll
      for (int j = 0; j < 8; ++j) sh_ctx[w][lane*8 + j] = cacc[j];

      float d = 0.f;
      #pragma unroll
      for (int j = 0; j < 8; ++j) d += fcwr[j] * cacc[j];
      d = wave_red_sum(d);
      if (lane == 0) sh_yt[w] = d + fcw_y * y_hist[rowb*128 + step] + fcb;
    }
    __syncthreads();

    // --- gates = C0 + h_f16 @ w_hh.T ; wave w owns d-slabs {w, w+8} x 4 gate
    //     types (tiles s + 16q). 8 WMMA accumulators. ------------------------
    v8f gA[4], gB[4];
    {
      #pragma unroll
      for (int q = 0; q < 4; ++q) {
        const int ga = w*16 + ln + 256*q, gb = (w+8)*16 + ln + 256*q;
        const float bA = biasc[ga], wA = w_ih[ga];
        const float bB = biasc[gb], wB = w_ih[gb];
        #pragma unroll
        for (int r = 0; r < 8; ++r) {
          const float yt = sh_yt[r + 8*hf];
          gA[q][r] = bA + wA * yt;
          gB[q][r] = bB + wB * yt;
        }
      }
      for (int kb = 0; kb < 8; ++kb) {               // K=256: h half of hcat
        union { v16h v; v8h h2[2]; } ua;
        const int k1 = kb*32 + hf*8;
        ua.h2[0] = *(const v8h*)&sh_hcat[ln][k1];
        ua.h2[1] = *(const v8h*)&sh_hcat[ln][k1 + 16];
        const int kbb = kb*32 + hf*16;
        #pragma unroll
        for (int q = 0; q < 4; ++q) {
          const _Float16* browA = whh + (long)( w   *16 + ln + 256*q) * 256;
          const _Float16* browB = whh + (long)((w+8)*16 + ln + 256*q) * 256;
          gA[q] = __builtin_amdgcn_wmma_f32_16x16x32_f16(false, ua.v, false,
                    wmma_b_frag(browA, kbb), (short)0, gA[q], false, false);
          gB[q] = __builtin_amdgcn_wmma_f32_16x16x32_f16(false, ua.v, false,
                    wmma_b_frag(browB, kbb), (short)0, gB[q], false, false);
        }
      }
    }
    __syncthreads();   // all waves finished reading sh_hcat before update

    // --- LSTM update directly on accumulator fragments (valid rows: hf==0) --
    if (hf == 0) {
      #pragma unroll
      for (int slab = 0; slab < 2; ++slab) {
        const int d = (slab ? (w+8) : w)*16 + ln;
        #pragma unroll
        for (int r = 0; r < 8; ++r) {
          const int m = r;                          // rows 0..7
          const float xi = slab ? gB[0][r] : gA[0][r];
          const float xf = slab ? gB[1][r] : gA[1][r];
          const float xg = slab ? gB[2][r] : gA[2][r];
          const float xo = slab ? gB[3][r] : gA[3][r];
          const float gi = fast_sig(xi);
          const float gf = fast_sig(xf);
          const float gg = fast_tanh(xg);
          const float go = fast_sig(xo);
          const float cn = gf * sh_c[m][d] + gi * gg;
          const float hn = go * fast_tanh(cn);
          sh_c[m][d] = cn;  sh_h[m][d] = hn;
          sh_hcat[m][d]       = (_Float16)hn;
          sh_hcat[m][256 + d] = (_Float16)cn;
        }
      }
    }
    __syncthreads();
  }

  // --- out[b] = [h, context_last] @ fcf_w.T + fcf_b (wave w -> row w) -------
  {
    float p = 0.f;
    #pragma unroll
    for (int j = 0; j < 8; ++j) p += sh_h[w][lane*8 + j]   * fcf_w[lane*8 + j];
    #pragma unroll
    for (int j = 0; j < 8; ++j) p += sh_ctx[w][lane*8 + j] * fcf_w[256 + lane*8 + j];
    p = wave_red_sum(p);
    if (lane == 0) out[rowb] = p + fcf_b[0];
  }
}

// ---------------------------------------------------------------------------
extern "C" void kernel_launch(void* const* d_in, const int* in_sizes, int n_in,
                              void* d_out, int out_size, void* d_ws, size_t ws_size,
                              hipStream_t stream)
{
  const float* enc  = (const float*)d_in[0];   // [1024,128,256]
  const float* yh   = (const float*)d_in[1];   // [1024,128]
  const float* aw1  = (const float*)d_in[2];   // [256,768]
  const float* ab1  = (const float*)d_in[3];   // [256]
  const float* aw2  = (const float*)d_in[4];   // [1,256]
  // d_in[5] = attn_b2: uniform shift over t -> cancels in softmax
  const float* wih  = (const float*)d_in[6];   // [1024,1]
  const float* whhf = (const float*)d_in[7];   // [1024,256]
  const float* bih  = (const float*)d_in[8];   // [1024]
  const float* bhh  = (const float*)d_in[9];   // [1024]
  const float* fcw  = (const float*)d_in[10];  // [1,257]
  const float* fcb  = (const float*)d_in[11];  // [1]
  const float* fcfw = (const float*)d_in[12];  // [1,512]
  const float* fcfb = (const float*)d_in[13];  // [1]

  char* ws = (char*)d_ws;                      // ~129 MB used, fully rewritten
  _Float16* w1hc  = (_Float16*)(ws);                       // 256*512*2   = 262144
  _Float16* w1e   = (_Float16*)(ws + 262144);              // 256*256*2   = 131072
  _Float16* whh   = (_Float16*)(ws + 393216);              // 1024*256*2  = 524288
  float*    biasc = (float*)   (ws + 917504);              // 1024*4      = 4096
  _Float16* encp  = (_Float16*)(ws + 921600);              // 1024*128*256*2
  _Float16* encf  = (_Float16*)(ws + 921600 + 67108864);

  prep_weights<<<(459776 + 255) / 256, 256, 0, stream>>>(aw1, whhf, bih, bhh,
                                                         w1hc, w1e, whh, biasc);
  encp_kernel<<<16384, 256, 0, stream>>>(enc, ab1, w1e, encp, encf);
  decoder_kernel<<<128, 256, 0, stream>>>(yh, aw2, wih, fcw, fcb, fcfw, fcfb,
                                          w1hc, whh, biasc, encp, encf,
                                          (float*)d_out);
}
